// LinformerFasterMultiHeadAttention_91242285236977
// MI455X (gfx1250) — matrix operand order repaired and checked by reading, compiled, not run on hardware
//
#include <hip/hip_runtime.h>

// ---------------------------------------------------------------------------
// Linformer MHA for MI455X (gfx1250, wave32, WMMA).
// All matmuls routed through v_wmma_f32_16x16x32_bf16 (fp32 accumulate).
// GEMM staging uses GLOBAL_LOAD_ASYNC_TO_LDS (ASYNCcnt) + double-buffered LDS
// when the toolchain exposes the builtin; falls back to sync loads otherwise.
// ---------------------------------------------------------------------------

#define B_   4
#define L_   4096
#define H_   1024
#define NH_  16
#define KP_  256     // Linformer projection length
#define DH_  64      // head dim

typedef __attribute__((ext_vector_type(16))) __bf16 v16bf;
typedef __attribute__((ext_vector_type(8)))  __bf16 v8bf;
typedef __attribute__((ext_vector_type(8)))  float  v8f;
typedef __attribute__((ext_vector_type(4)))  int    v4i;

__device__ inline __bf16 f2bf(float f) {
  union { float f; unsigned u; } v; v.f = f;
  unsigned r = v.u + 0x7FFFu + ((v.u >> 16) & 1u);   // round-to-nearest-even
  unsigned short h = (unsigned short)(r >> 16);
  __bf16 b;
  __builtin_memcpy(&b, &h, sizeof(b));
  return b;
}

__device__ inline v16bf cat16(v8bf lo, v8bf hi) {
  return __builtin_shufflevector(lo, hi, 0,1,2,3,4,5,6,7,8,9,10,11,12,13,14,15);
}

__device__ inline v8f v8zero() {
  v8f z = {0.f,0.f,0.f,0.f,0.f,0.f,0.f,0.f};
  return z;
}

// --- CDNA5 async global->LDS copy (16B per lane), with sync fallback -------
#if __has_builtin(__builtin_amdgcn_global_load_async_to_lds_b128)
#define HAVE_ASYNC_LDS 1
#else
#define HAVE_ASYNC_LDS 0
#endif

typedef __attribute__((address_space(1))) v4i* gv4i_ptr;   // global (prints __device__)
typedef __attribute__((address_space(3))) v4i* lv4i_ptr;   // LDS    (prints __shared__)

__device__ inline void async_ld_b128(const __bf16* g, __bf16* l) {
#if HAVE_ASYNC_LDS
  gv4i_ptr gp = (gv4i_ptr)const_cast<__bf16*>(g);
  lv4i_ptr lp = (lv4i_ptr)l;
  __builtin_amdgcn_global_load_async_to_lds_b128(gp, lp, 0, 0);
#else
  *(v8bf*)l = *(const v8bf*)g;
#endif
}

__device__ inline void wait_async_lds() {
#if HAVE_ASYNC_LDS
#if __has_builtin(__builtin_amdgcn_s_wait_asynccnt)
  __builtin_amdgcn_s_wait_asynccnt(0);
#else
  asm volatile("s_wait_asynccnt 0x0" ::: "memory");
#endif
#endif
}

// ---------------------------------------------------------------------------
// fp32 -> bf16 conversion (grid-stride)
// ---------------------------------------------------------------------------
__global__ __launch_bounds__(256)
void cvt_f32_bf16(const float* __restrict__ src, __bf16* __restrict__ dst, int n) {
  int i = blockIdx.x * blockDim.x + threadIdx.x;
  int stride = gridDim.x * blockDim.x;
  for (; i < n; i += stride) dst[i] = f2bf(src[i]);
}

// ---------------------------------------------------------------------------
// Generic LDS-tiled bf16 GEMM, C(M,N) = A(M,Kd) @ B(Kd,N) [+ bias]
//   BM=128, BN=64, BK=32. 256 threads = 8 waves, each wave owns 32x32
//   (2x2 WMMA accumulators). Double-buffered LDS; A staged via async-to-LDS.
// MODE 0: flat, bf16 out + bias        (qkv = x @ W_qkv + b_qkv)
// MODE 1: batched z=(b,h), B = k-slice of qkv, bf16 out [z][m*DH+n]   (K_proj)
// MODE 2: batched z=(b,h), B = v-slice of qkv, bf16 out TRANSPOSED
//         [z][n*KP+m]  -> V_proj^T, so attention B-fragments are contiguous
// MODE 3: flat, f32 out + bias         (out = attn @ W_out + b_out)
// ---------------------------------------------------------------------------
#define BM 128
#define BN 64
#define BK 32

template<int MODE>
__global__ __launch_bounds__(256)
void gemm_bf16(const __bf16* __restrict__ A, const __bf16* __restrict__ Bm,
               float* __restrict__ Cf, __bf16* __restrict__ Cb,
               const float* __restrict__ bias,
               int M, int N, int Kd, int lda, int ldb, int ldc, int qoff)
{
  __shared__ __attribute__((aligned(16))) __bf16 As[2][BM][BK];  // 2 x 8 KB
  __shared__ __attribute__((aligned(16))) __bf16 Bt[2][BN][BK];  // 2 x 4 KB, B^T: [n][k]

  const int t  = threadIdx.x;
  const int bm = blockIdx.x * BM;
  const int bn = blockIdx.y * BN;
  const int z  = blockIdx.z;

  const __bf16* Abase = A;
  const __bf16* Bbase = Bm;
  if (MODE == 1 || MODE == 2) {
    const int bb = z >> 4, hh = z & 15;                 // z = b*NH + h
    Bbase = Bm + ((size_t)bb * L_) * (3 * H_) + qoff + hh * DH_;
  }

  const int w    = t >> 5;            // wave id 0..7
  const int lane = t & 31;
  const int wm   = w & 3;             // wave M position (4)
  const int wn   = w >> 2;            // wave N position (2)
  const int lm   = lane & 15;
  const int half = lane >> 4;

  v8f acc[2][2];
  #pragma unroll
  for (int i = 0; i < 2; i++)
    #pragma unroll
    for (int j = 0; j < 2; j++) acc[i][j] = v8zero();

  // staging assignments
  const int ar = t >> 1;              // A row 0..127
  const int ac = (t & 1) << 4;        // A col 0 or 16
  const int bk = t >> 3;              // B k-row 0..31
  const int bc = (t & 7) << 3;        // B col 0..56 step 8

  const __bf16* aRowBase = Abase + (size_t)(bm + ar) * lda + ac;
  const __bf16* bRowBase = Bbase + (size_t)bk * ldb + bn + bc;

  auto stage = [&](int buf, int k0) {
    // A tile: per-lane 2 x 16B async copies into LDS (row-major)
    const __bf16* ap = aRowBase + k0;
    async_ld_b128(ap,     &As[buf][ar][ac]);
    async_ld_b128(ap + 8, &As[buf][ar][ac + 8]);
    // B tile: transpose scatter through VGPRs (async cannot scatter)
    const __bf16* bp = bRowBase + (size_t)k0 * ldb;
    v8bf bv = *(const v8bf*)bp;
    #pragma unroll
    for (int i = 0; i < 8; i++) Bt[buf][bc + i][bk] = bv[i];
  };

  // prologue: fill buffer 0
  stage(0, 0);
  wait_async_lds();
  __syncthreads();

  int cur = 0;
  for (int k0 = 0; k0 < Kd; k0 += BK) {
    if (k0 + BK < Kd) {
      stage(cur ^ 1, k0 + BK);                 // overlap next-tile staging
      __builtin_prefetch(aRowBase + k0 + 2 * BK, 0, 3);  // global_prefetch_b8
    }

    // --- 2x2 WMMA tiles per wave out of buffer `cur` ---
    #pragma unroll
    for (int mt = 0; mt < 2; mt++) {
      // A frag: lane m = lm; elem e -> K = half*16 + e (contiguous)
      const __bf16* arow = &As[cur][wm * 32 + mt * 16 + lm][half * 16];
      v16bf af = cat16(*(const v8bf*)arow, *(const v8bf*)(arow + 8));
      #pragma unroll
      for (int nt = 0; nt < 2; nt++) {
        // B frag: lane n = lm; elem e -> K = half*16 + e (contiguous in Bt row)
        const __bf16* brow = &Bt[cur][wn * 32 + nt * 16 + lm][half * 16];
        v16bf bf = cat16(*(const v8bf*)brow, *(const v8bf*)(brow + 8));
        acc[mt][nt] = __builtin_amdgcn_wmma_f32_16x16x32_bf16(
            false, af, false, bf, (short)0, acc[mt][nt], false, false);
      }
    }

    wait_async_lds();     // own async copies for buf^1 landed
    __syncthreads();      // all waves' staging visible, all reads of `cur` done
    cur ^= 1;
  }

  // --- epilogue (D layout: elem r -> row m = r + half*8, col n = lm) ---
  #pragma unroll
  for (int mt = 0; mt < 2; mt++) {
    #pragma unroll
    for (int nt = 0; nt < 2; nt++) {
      #pragma unroll
      for (int r = 0; r < 8; r++) {
        const int row = bm + wm * 32 + mt * 16 + half * 8 + r;
        const int col = bn + wn * 32 + nt * 16 + lm;
        const float v = acc[mt][nt][r];
        if (MODE == 0) {
          Cb[(size_t)row * ldc + col] = f2bf(v + bias[col]);
        } else if (MODE == 1) {
          Cb[(size_t)z * (KP_ * DH_) + (size_t)row * DH_ + col] = f2bf(v);
        } else if (MODE == 2) {
          Cb[(size_t)z * (KP_ * DH_) + (size_t)col * KP_ + row] = f2bf(v);
        } else {
          Cf[(size_t)row * ldc + col] = v + bias[col];
        }
      }
    }
  }
}

// ---------------------------------------------------------------------------
// Attention: per (b,h), per 16-row block of L (one wave each, 4 waves/block):
//   scores(16x256) = q(16x64) @ Kproj^T  (16 WMMA tiles, 2 k-steps each)
//   register-resident softmax (shfl_xor over the 16-lane half-wave that owns
//   each output row in the C/D layout), probs -> LDS in A-fragment layout,
//   attn(16x64) = probs(16x256) @ Vproj  (via VT[dh][k], 4 tiles x 8 k-steps)
// ---------------------------------------------------------------------------
__global__ __launch_bounds__(128)
void attn_kernel(const __bf16* __restrict__ qkv, const __bf16* __restrict__ Kp,
                 const __bf16* __restrict__ VT, __bf16* __restrict__ attnOut)
{
  __shared__ __attribute__((aligned(16))) __bf16 probs[4][16][KP_];  // 32 KB

  const int z    = blockIdx.y;
  const int bb   = z >> 4, hh = z & 15;
  const int w    = threadIdx.x >> 5;
  const int lane = threadIdx.x & 31;
  const int lm   = lane & 15;
  const int half = lane >> 4;
  const int l0   = blockIdx.x * 64 + w * 16;

  // q fragments (A-matrix), DH=64 -> 2 k-steps of 32
  const __bf16* qrow = qkv + ((size_t)bb * L_ + (size_t)(l0 + lm)) * (3 * H_) + hh * DH_;
  v16bf qf[2];
  #pragma unroll
  for (int s = 0; s < 2; s++) {
    const __bf16* p = qrow + s * 32 + half * 16;
    qf[s] = cat16(*(const v8bf*)p, *(const v8bf*)(p + 8));
  }

  // scores = q @ Kproj^T, scaled by 1/sqrt(DH) = 0.125
  const __bf16* kp = Kp + (size_t)z * (KP_ * DH_);
  v8f sc[16];
  #pragma unroll
  for (int j = 0; j < 16; j++) {
    v8f c = v8zero();
    #pragma unroll
    for (int s = 0; s < 2; s++) {
      // B frag: B[k=dh][n=kidx] = Kproj[kidx][dh] -> contiguous in dh
      const __bf16* p = kp + (size_t)(j * 16 + lm) * DH_ + s * 32 + half * 16;
      v16bf bf = cat16(*(const v8bf*)p, *(const v8bf*)(p + 8));
      c = __builtin_amdgcn_wmma_f32_16x16x32_bf16(false, qf[s], false, bf,
                                                  (short)0, c, false, false);
    }
    #pragma unroll
    for (int r = 0; r < 8; r++) c[r] *= 0.125f;
    sc[j] = c;
  }

  // softmax over K (row m = r + half*8 lives in the 16 lanes of one half;
  // xor masks 1,2,4,8 keep the half bit fixed)
  #pragma unroll
  for (int r = 0; r < 8; r++) {
    float mx = -3.0e38f;
    #pragma unroll
    for (int j = 0; j < 16; j++) mx = fmaxf(mx, sc[j][r]);
    #pragma unroll
    for (int d = 1; d < 16; d <<= 1) mx = fmaxf(mx, __shfl_xor(mx, d, 32));
    float sum = 0.f;
    #pragma unroll
    for (int j = 0; j < 16; j++) {
      float e = __expf(sc[j][r] - mx);
      sc[j][r] = e;
      sum += e;
    }
    #pragma unroll
    for (int d = 1; d < 16; d <<= 1) sum += __shfl_xor(sum, d, 32);
    const float inv = 1.0f / sum;
    #pragma unroll
    for (int j = 0; j < 16; j++)
      probs[w][half * 8 + r][j * 16 + lm] = f2bf(sc[j][r] * inv);
  }
  __syncthreads();

  // attn = probs @ Vproj   (B frags from VT[dh][k]: contiguous in k)
  const __bf16* vt = VT + (size_t)z * (KP_ * DH_);
  #pragma unroll
  for (int nt = 0; nt < 4; nt++) {
    v8f c = v8zero();
    #pragma unroll
    for (int s = 0; s < 8; s++) {
      const __bf16* ap = &probs[w][lm][s * 32 + half * 16];
      v16bf af = cat16(*(const v8bf*)ap, *(const v8bf*)(ap + 8));
      const __bf16* bp = vt + (size_t)(nt * 16 + lm) * KP_ + s * 32 + half * 16;
      v16bf bf = cat16(*(const v8bf*)bp, *(const v8bf*)(bp + 8));
      c = __builtin_amdgcn_wmma_f32_16x16x32_bf16(false, af, false, bf,
                                                  (short)0, c, false, false);
    }
    #pragma unroll
    for (int r = 0; r < 8; r++) {
      const size_t row = (size_t)bb * L_ + (size_t)(l0 + half * 8 + r);
      attnOut[row * H_ + hh * DH_ + nt * 16 + lm] = f2bf(c[r]);
    }
  }
}

// ---------------------------------------------------------------------------
// Host-side pipeline
// ---------------------------------------------------------------------------
extern "C" void kernel_launch(void* const* d_in, const int* in_sizes, int n_in,
                              void* d_out, int out_size, void* d_ws, size_t ws_size,
                              hipStream_t stream) {
  (void)in_sizes; (void)n_in; (void)out_size; (void)ws_size;

  const float* x     = (const float*)d_in[0];
  const float* W_qkv = (const float*)d_in[1];
  const float* b_qkv = (const float*)d_in[2];
  const float* E_K   = (const float*)d_in[3];
  const float* E_V   = (const float*)d_in[4];
  const float* W_out = (const float*)d_in[5];
  const float* b_out = (const float*)d_in[6];
  float* out = (float*)d_out;

  const size_t nX   = (size_t)B_ * L_ * H_;          // 16M
  const size_t nWq  = (size_t)H_ * 3 * H_;           // 3M
  const size_t nE   = (size_t)KP_ * L_;              // 1M
  const size_t nWo  = (size_t)H_ * H_;               // 1M
  const size_t nQKV = (size_t)B_ * L_ * 3 * H_;      // 48M
  const size_t nP   = (size_t)B_ * NH_ * KP_ * DH_;  // 1M

  __bf16* p    = (__bf16*)d_ws;
  __bf16* xb   = p; p += nX;
  __bf16* wqb  = p; p += nWq;
  __bf16* ekb  = p; p += nE;
  __bf16* evb  = p; p += nE;
  __bf16* wob  = p; p += nWo;
  __bf16* qkvb = p; p += nQKV;
  __bf16* kpb  = p; p += nP;
  __bf16* vtb  = p; p += nP;
  __bf16* attb = p; p += nX;

  // 1) fp32 -> bf16 operand conversion
  cvt_f32_bf16<<<4096, 256, 0, stream>>>(x,     xb,  (int)nX);
  cvt_f32_bf16<<<2048, 256, 0, stream>>>(W_qkv, wqb, (int)nWq);
  cvt_f32_bf16<<<1024, 256, 0, stream>>>(E_K,   ekb, (int)nE);
  cvt_f32_bf16<<<1024, 256, 0, stream>>>(E_V,   evb, (int)nE);
  cvt_f32_bf16<<<1024, 256, 0, stream>>>(W_out, wob, (int)nWo);

  // 2) qkv = x @ W_qkv + b_qkv     (16384 x 3072 x 1024)
  gemm_bf16<0><<<dim3(16384 / BM, (3 * H_) / BN, 1), 256, 0, stream>>>(
      xb, wqb, nullptr, qkvb, b_qkv,
      B_ * L_, 3 * H_, H_, H_, 3 * H_, 3 * H_, 0);

  // 3) K_proj[b,h] = E_K @ k_slice   (256 x 64 x 4096, batched over 64 heads)
  gemm_bf16<1><<<dim3(KP_ / BM, DH_ / BN, B_ * NH_), 256, 0, stream>>>(
      ekb, qkvb, nullptr, kpb, nullptr,
      KP_, DH_, L_, L_, 3 * H_, DH_, H_);

  //    V_proj^T[b,h] = (E_V @ v_slice)^T  (transposed epilogue -> VT[dh][k])
  gemm_bf16<2><<<dim3(KP_ / BM, DH_ / BN, B_ * NH_), 256, 0, stream>>>(
      evb, qkvb, nullptr, vtb, nullptr,
      KP_, DH_, L_, L_, 3 * H_, KP_, 2 * H_);

  // 4) attention (scores -> softmax -> probs @ V_proj)
  attn_kernel<<<dim3(L_ / 64, B_ * NH_), 128, 0, stream>>>(qkvb, kpb, vtb, attb);

  // 5) out = attn @ W_out + b_out  (16384 x 1024 x 1024, fp32 output)
  gemm_bf16<3><<<dim3(16384 / BM, H_ / BN, 1), 256, 0, stream>>>(
      attb, wob, out, nullptr, b_out,
      B_ * L_, H_, H_, H_, H_, H_, 0);
}
